// MoDBlock_7378753814622
// MI455X (gfx1250) — compile-verified, matrix operand order checked by reading, not run on hardware
//
#include <hip/hip_runtime.h>
#include <hip/hip_bf16.h>
#include <math.h>

// ---------------------------------------------------------------------------
// MoD block for MI455X (gfx1250, wave32). bf16 WMMA (v_wmma_f32_16x16x32_bf16)
// for all GEMMs and attention; flash-style attention (no score materialization);
// async Global->LDS DMA staging for WMMA B operands (coalesced b128 copies,
// ASYNCcnt-tracked) instead of per-lane scattered u16 global loads;
// bitonic top-k; fused RMSNorm+RoPE; gated combine + scatter; BCE loss.
// Problem constants (from reference): B=2, T=2048, C=512, E=2, NH=8, hs=64, k=1024.
// ---------------------------------------------------------------------------

#define BB   2
#define TT   2048
#define CC   512
#define EE   2
#define NHH  8
#define HS   64
#define KSEL 1024

typedef __bf16 v16bf __attribute__((ext_vector_type(16)));
typedef float  v8f   __attribute__((ext_vector_type(8)));

__device__ __forceinline__ v8f wmma_bf16(v16bf a, v16bf b, v8f c) {
  // D = A(16x32 bf16) * B(32x16 bf16) + C(16x16 f32)
  return __builtin_amdgcn_wmma_f32_16x16x32_bf16(false, a, false, b, (short)0, c,
                                                 false, false);
}

// Async Global->LDS 16-byte copy (per lane), tracked by ASYNCcnt.
// LDS dest address (byte offset) in a VGPR, 64-bit global address in a VGPR pair.
__device__ __forceinline__ void async_load_b128(const void* gaddr, unsigned lds_off) {
  unsigned long long ga = (unsigned long long)(size_t)gaddr;
  asm volatile("global_load_async_to_lds_b128 %0, %1, off"
               :
               : "v"(lds_off), "v"(ga)
               : "memory");
}
__device__ __forceinline__ void wait_async0() {
  asm volatile("s_wait_asynccnt 0" ::: "memory");
}
// low 32 bits of a generic pointer to LDS == LDS byte address
__device__ __forceinline__ unsigned lds_addr_of(const void* p) {
  return (unsigned)(size_t)p;
}

// ---------------------------------------------------------------------------
// f32 -> bf16 conversion (weights)
// ---------------------------------------------------------------------------
__global__ __launch_bounds__(256) void tobf16_kernel(const float* __restrict__ s,
                                                     __bf16* __restrict__ d, int n) {
  int i = blockIdx.x * 256 + threadIdx.x;
  if (i < n) d[i] = (__bf16)s[i];
}

// ---------------------------------------------------------------------------
// Router + predictor logits (one wave per token), zero the BCE target mask
// ---------------------------------------------------------------------------
__global__ __launch_bounds__(256) void router_kernel(const float* __restrict__ x,
                                                     const float* __restrict__ wr,
                                                     const float* __restrict__ wp,
                                                     float* __restrict__ logits,
                                                     float* __restrict__ pl,
                                                     float* __restrict__ maskf) {
  int token = blockIdx.x * 8 + (threadIdx.x >> 5);
  int lane  = threadIdx.x & 31;
  const float* xr = x + (size_t)token * CC;
  float s0 = 0.f, s1 = 0.f;
  for (int c = lane; c < CC; c += 32) {
    float v = xr[c];
    s0 += v * wr[c];
    s1 += v * wp[c];
  }
  for (int m = 16; m > 0; m >>= 1) {
    s0 += __shfl_xor(s0, m);
    s1 += __shfl_xor(s1, m);
  }
  if (lane == 0) {
    logits[token] = s0;
    pl[token]     = s1;
    maskf[token]  = 0.0f;
  }
}

// ---------------------------------------------------------------------------
// Per-batch top-k: bitonic sort of 2048 packed keys (value desc, index asc on
// ties -> matches jax.lax.top_k), then ascending re-sort of the 1024 winning
// indices (matches argsort(idx) temporal restore). Writes idx, sigmoid weight,
// and the BCE target mask.
// ---------------------------------------------------------------------------
__global__ __launch_bounds__(1024) void topk_kernel(const float* __restrict__ logits,
                                                    int* __restrict__ idx,
                                                    float* __restrict__ wsel,
                                                    float* __restrict__ maskf) {
  int b = blockIdx.x;
  __shared__ unsigned long long keys[TT];
  __shared__ int sidx[KSEL];
  const float* lg = logits + (size_t)b * TT;

  for (int i = threadIdx.x; i < TT; i += 1024) {
    unsigned u = __float_as_uint(lg[i]);
    unsigned m = (u & 0x80000000u) ? ~u : (u | 0x80000000u);  // monotonic map
    keys[i] = ((unsigned long long)m << 32) | (unsigned)(TT - 1 - i);
  }
  __syncthreads();

  // descending bitonic sort over 2048 keys
  for (unsigned sz = 2; sz <= TT; sz <<= 1) {
    for (unsigned st = sz >> 1; st > 0; st >>= 1) {
      for (unsigned i = threadIdx.x; i < TT; i += 1024) {
        unsigned p = i ^ st;
        if (p > i) {
          bool up = ((i & sz) == 0);
          unsigned long long a = keys[i], c = keys[p];
          bool sw = up ? (a < c) : (a > c);  // descending overall
          if (sw) { keys[i] = c; keys[p] = a; }
        }
      }
      __syncthreads();
    }
  }

  sidx[threadIdx.x] = (TT - 1) - (int)(keys[threadIdx.x] & 0xFFFFFFFFu);
  __syncthreads();

  // ascending bitonic sort of the 1024 selected indices
  for (unsigned sz = 2; sz <= KSEL; sz <<= 1) {
    for (unsigned st = sz >> 1; st > 0; st >>= 1) {
      unsigned i = threadIdx.x;
      unsigned p = i ^ st;
      if (p > i) {
        bool up = ((i & sz) == 0);
        int a = sidx[i], c = sidx[p];
        bool sw = up ? (a > c) : (a < c);
        if (sw) { sidx[i] = c; sidx[p] = a; }
      }
      __syncthreads();
    }
  }

  int t = sidx[threadIdx.x];
  idx[b * KSEL + threadIdx.x]  = t;
  float lv = lg[t];
  wsel[b * KSEL + threadIdx.x] = 1.0f / (1.0f + expf(-lv));
  maskf[b * TT + t]            = 1.0f;
}

// ---------------------------------------------------------------------------
// Gather selected tokens to bf16 (GEMM A operand)
// ---------------------------------------------------------------------------
__global__ __launch_bounds__(128) void gather_kernel(const float* __restrict__ x,
                                                     const int* __restrict__ idx,
                                                     __bf16* __restrict__ sel) {
  int row = blockIdx.x;  // b*KSEL + j
  int b   = row >> 10;
  int t   = idx[row];
  const float* src = x + ((size_t)b * TT + t) * CC;
  __bf16* dst = sel + (size_t)row * CC;
  for (int c = threadIdx.x; c < CC; c += 128) dst[c] = (__bf16)src[c];
}

// ---------------------------------------------------------------------------
// Generic bf16 GEMM: C[M,N] f32 = A[M,K] @ B[K,N], row-major, one wave per
// 16x16 output tile, K in steps of 32 via v_wmma_f32_16x16x32_bf16.
// A operand: direct b128 global loads (rows contiguous).
// B operand: 32x16 tile staged to LDS with async b128 DMA (coalesced 32B/row),
// then fine-grained gather from LDS per ISA 7.12.2 layout.
// ---------------------------------------------------------------------------
__global__ __launch_bounds__(256) void gemm_bf16_kernel(const __bf16* __restrict__ A,
                                                        const __bf16* __restrict__ B,
                                                        float* __restrict__ C,
                                                        int M, int N, int K) {
  __shared__ __bf16 btile[8][32 * 16];  // per-wave slice: 1 KB
  int w      = threadIdx.x >> 5;
  int wave   = blockIdx.x * 8 + w;
  int ntiles = N >> 4;
  int tm = wave / ntiles;
  int tn = wave - tm * ntiles;
  if (tm >= (M >> 4)) return;  // wave-uniform guard

  int lane = threadIdx.x & 31;
  int half = lane >> 4;
  int lr   = lane & 15;

  const __bf16* arow  = A + (size_t)(tm * 16 + lr) * K;
  const unsigned bt0  = lds_addr_of(&btile[w][0]);
  const __bf16* bcol  = B + (size_t)tn * 16;
  v8f acc = {};

  for (int k0 = 0; k0 < K; k0 += 32) {
    // --- async-stage B rows k0..k0+31 (16 cols = 32 B each) into LDS ---
    {
      const __bf16* bg = bcol + (size_t)(k0 + lane) * N;
      unsigned dst = bt0 + lane * 32;
      async_load_b128(bg, dst);
      async_load_b128(bg + 8, dst + 16);
    }

    int base = k0 + half * 8;
    __builtin_prefetch(arow + base + 128, 0, 1);  // global_prefetch_b8
    v16bf a, bb;
#pragma unroll
    for (int i = 0; i < 8; i++) {
      a[i]     = arow[base + i];        // K in {h*8 .. h*8+7}
      a[i + 8] = arow[base + 16 + i];   // K in {16+h*8 .. 16+h*8+7}
    }

    wait_async0();
    int lb = half * 8;
#pragma unroll
    for (int i = 0; i < 8; i++) {
      bb[i]     = btile[w][(lb + i) * 16 + lr];
      bb[i + 8] = btile[w][(lb + 16 + i) * 16 + lr];
    }
    acc = wmma_bf16(a, bb, acc);
  }

#pragma unroll
  for (int r = 0; r < 8; r++)
    C[(size_t)(tm * 16 + r + half * 8) * N + tn * 16 + lr] = acc[r];
}

// ---------------------------------------------------------------------------
// RoPE angle helper: rot[d] = pos * 10000^((-4*(d%32) - 1)/64)
// ---------------------------------------------------------------------------
__device__ __forceinline__ float rope_angle(int pos, int d) {
  float inv = powf(10000.0f, (-4.0f * (float)(d & 31) - 1.0f) * (1.0f / 64.0f));
  return (float)pos * inv;
}

// ---------------------------------------------------------------------------
// q post: RMSNorm over C, RoPE per head, write bf16 head-major [B,E,NH,k,hs]
// ---------------------------------------------------------------------------
__global__ __launch_bounds__(128) void q_post_kernel(const float* __restrict__ qf,
                                                     const float* __restrict__ lnw,
                                                     __bf16* __restrict__ qh) {
  int id = blockIdx.x;           // b*E*KSEL + e*KSEL + j
  int j  = id & (KSEL - 1);
  int e  = (id >> 10) & 1;
  int b  = id >> 11;
  const float* row = qf + ((size_t)(b * KSEL + j)) * (EE * CC) + e * CC;

  __shared__ float red[128];
  __shared__ float rowbuf[CC];
  float ss = 0.f;
  for (int c = threadIdx.x; c < CC; c += 128) {
    float v = row[c];
    rowbuf[c] = v;
    ss += v * v;
  }
  red[threadIdx.x] = ss;
  __syncthreads();
  for (int s = 64; s > 0; s >>= 1) {
    if (threadIdx.x < s) red[threadIdx.x] += red[threadIdx.x + s];
    __syncthreads();
  }
  float rms = rsqrtf(red[0] * (1.0f / CC) + 1e-6f);

  for (int c = threadIdx.x; c < CC; c += 128) {
    int h = c >> 6, d = c & 63;
    float u  = rowbuf[c] * rms * lnw[c];
    float up = rowbuf[c ^ 1] * rms * lnw[c ^ 1];
    float rotv = (d & 1) ? up : -up;           // rotate: (x0,x1)->(-x1,x0)
    float ang  = rope_angle(j, d);
    float o = u * cosf(ang) + rotv * sinf(ang);
    qh[(((size_t)((b * EE + e) * NHH + h)) * KSEL + j) * HS + d] = (__bf16)o;
  }
}

// ---------------------------------------------------------------------------
// kv post: K gets RMSNorm+RoPE, V passthrough; write bf16 [B,NH,k,hs]
// ---------------------------------------------------------------------------
__global__ __launch_bounds__(128) void kv_post_kernel(const float* __restrict__ kvf,
                                                      const float* __restrict__ lnw,
                                                      __bf16* __restrict__ kh,
                                                      __bf16* __restrict__ vh) {
  int row = blockIdx.x;  // b*KSEL + j
  int b = row >> 10, j = row & (KSEL - 1);
  const float* rp = kvf + (size_t)row * (2 * CC);

  __shared__ float red[128];
  __shared__ float rowbuf[CC];
  float ss = 0.f;
  for (int c = threadIdx.x; c < CC; c += 128) {
    float v = rp[c];
    rowbuf[c] = v;
    ss += v * v;
  }
  red[threadIdx.x] = ss;
  __syncthreads();
  for (int s = 64; s > 0; s >>= 1) {
    if (threadIdx.x < s) red[threadIdx.x] += red[threadIdx.x + s];
    __syncthreads();
  }
  float rms = rsqrtf(red[0] * (1.0f / CC) + 1e-6f);

  for (int c = threadIdx.x; c < CC; c += 128) {
    int h = c >> 6, d = c & 63;
    float u  = rowbuf[c] * rms * lnw[c];
    float up = rowbuf[c ^ 1] * rms * lnw[c ^ 1];
    float rotv = (d & 1) ? up : -up;
    float ang  = rope_angle(j, d);
    float o = u * cosf(ang) + rotv * sinf(ang);
    size_t hd = (((size_t)(b * NHH + h)) * KSEL + j) * HS + d;
    kh[hd] = (__bf16)o;
    vh[hd] = (__bf16)rp[CC + c];
  }
}

// ---------------------------------------------------------------------------
// Flash attention: one wave per 16-query block per (b,e,h).
// K/V 32-key tiles are async-DMA'd into LDS (16 x b128 per tile, coalesced),
// S = QK^T via 2 WMMAs per 16-key tile (hs=64 = 2x32); online softmax in
// registers (row stats via 16-lane shfl_xor reductions); P redistributed
// through a 1KB LDS tile into A-layout; PV via 4 WMMAs (K=32 keys/tile).
// ---------------------------------------------------------------------------
__global__ __launch_bounds__(32) void attn_kernel(const __bf16* __restrict__ qh,
                                                  const __bf16* __restrict__ kh,
                                                  const __bf16* __restrict__ vh,
                                                  __bf16* __restrict__ yatt) {
  int tile = blockIdx.x;
  int qb = tile & 63; tile >>= 6;    // query block (0..63)
  int h  = tile & 7;  tile >>= 3;
  int e  = tile & 1;  tile >>= 1;
  int b  = tile;

  int lane = threadIdx.x;
  int half = lane >> 4;
  int lr   = lane & 15;
  int q0   = qb * 16;

  const __bf16* Q  = qh + ((size_t)((b * EE + e) * NHH + h)) * KSEL * HS;
  const __bf16* Kk = kh + ((size_t)(b * NHH + h)) * KSEL * HS;
  const __bf16* Vv = vh + ((size_t)(b * NHH + h)) * KSEL * HS;

  __shared__ __bf16 ktile[32 * HS];   // 4 KB: keys j0..j0+31, row-major [32][64]
  __shared__ __bf16 vtile[32 * HS];   // 4 KB
  __shared__ __bf16 plds[16 * 32];    // 1 KB: P tile redistribution
  const unsigned klds = lds_addr_of(&ktile[0]);
  const unsigned vlds = lds_addr_of(&vtile[0]);

  // A operands: Q rows 16 x (hs 0..31) and 16 x (hs 32..63)
  v16bf aq0, aq1;
  {
    const __bf16* qr = Q + (size_t)(q0 + lr) * HS;
    int base = half * 8;
#pragma unroll
    for (int i = 0; i < 8; i++) {
      aq0[i]     = qr[base + i];
      aq0[i + 8] = qr[base + 16 + i];
      aq1[i]     = qr[32 + base + i];
      aq1[i + 8] = qr[32 + base + 16 + i];
    }
  }

  float m_i[8], l_i[8];
  v8f Oacc[4];
#pragma unroll
  for (int r = 0; r < 8; r++) { m_i[r] = -3.0e38f; l_i[r] = 0.f; }
#pragma unroll
  for (int c = 0; c < 4; c++) Oacc[c] = (v8f){};

  const float scale = 0.125f;  // 1/sqrt(64)
  int ntiles = (q0 + 16 + 31) >> 5;  // causal: keys 0 .. q0+15, 32 per tile

  for (int t32 = 0; t32 < ntiles; ++t32) {
    int j0 = t32 * 32;

    // --- async-stage K and V rows j0..j0+31 (128 B per row) into LDS ---
    {
      const __bf16* kg = Kk + (size_t)(j0 + lane) * HS;
      const __bf16* vg = Vv + (size_t)(j0 + lane) * HS;
      unsigned kd = klds + lane * 128;
      unsigned vd = vlds + lane * 128;
#pragma unroll
      for (int ch = 0; ch < 8; ch++) {
        async_load_b128(kg + ch * 8, kd + ch * 16);
        async_load_b128(vg + ch * 8, vd + ch * 16);
      }
      wait_async0();
    }

    v8f S0 = {}, S1 = {};
    {
      v16bf bk0, bk1, ck0, ck1;
      int base = half * 8;
      int r0 = lr * HS, r1 = (16 + lr) * HS;
#pragma unroll
      for (int i = 0; i < 8; i++) {
        bk0[i]     = ktile[r0 + base + i];       bk0[i + 8] = ktile[r0 + base + 16 + i];
        bk1[i]     = ktile[r0 + 32 + base + i];  bk1[i + 8] = ktile[r0 + 32 + base + 16 + i];
        ck0[i]     = ktile[r1 + base + i];       ck0[i + 8] = ktile[r1 + base + 16 + i];
        ck1[i]     = ktile[r1 + 32 + base + i];  ck1[i + 8] = ktile[r1 + 32 + base + 16 + i];
      }
      S0 = wmma_bf16(aq0, bk0, S0);
      S0 = wmma_bf16(aq1, bk1, S0);
      S1 = wmma_bf16(aq0, ck0, S1);
      S1 = wmma_bf16(aq1, ck1, S1);
    }

    int col0 = j0 + lr, col1 = col0 + 16;
    // causal mask + scale + online softmax stats
#pragma unroll
    for (int r = 0; r < 8; r++) {
      int row = q0 + r + half * 8;
      float s0 = (col0 <= row) ? S0[r] * scale : -3.0e38f;
      float s1 = (col1 <= row) ? S1[r] * scale : -3.0e38f;
      float t = fmaxf(s0, s1);
      t = fmaxf(t, __shfl_xor(t, 1));
      t = fmaxf(t, __shfl_xor(t, 2));
      t = fmaxf(t, __shfl_xor(t, 4));
      t = fmaxf(t, __shfl_xor(t, 8));
      float newm = fmaxf(m_i[r], t);
      float al = expf(m_i[r] - newm);
      float p0 = expf(s0 - newm);
      float p1 = expf(s1 - newm);
      S0[r] = p0;
      S1[r] = p1;
      float ps = p0 + p1;
      ps += __shfl_xor(ps, 1);
      ps += __shfl_xor(ps, 2);
      ps += __shfl_xor(ps, 4);
      ps += __shfl_xor(ps, 8);
      l_i[r] = l_i[r] * al + ps;
      m_i[r] = newm;
      Oacc[0][r] *= al; Oacc[1][r] *= al; Oacc[2][r] *= al; Oacc[3][r] *= al;
    }

    // redistribute P (C/D layout -> A layout) through LDS
    __syncthreads();
#pragma unroll
    for (int r = 0; r < 8; r++) {
      int row = r + half * 8;
      plds[row * 32 + lr]      = (__bf16)S0[r];
      plds[row * 32 + 16 + lr] = (__bf16)S1[r];
    }
    __syncthreads();

    v16bf ap;
    {
      int base = half * 8;
#pragma unroll
      for (int i = 0; i < 8; i++) {
        ap[i]     = plds[lr * 32 + base + i];
        ap[i + 8] = plds[lr * 32 + 16 + base + i];
      }
    }

    // PV: 4 hs-chunks of 16 columns, K = 32 keys (B operand from LDS V tile)
#pragma unroll
    for (int c = 0; c < 4; c++) {
      v16bf bv;
      int base = half * 8;
#pragma unroll
      for (int i = 0; i < 8; i++) {
        bv[i]     = vtile[(base + i) * HS + c * 16 + lr];
        bv[i + 8] = vtile[(16 + base + i) * HS + c * 16 + lr];
      }
      Oacc[c] = wmma_bf16(ap, bv, Oacc[c]);
    }
    __syncthreads();
  }

  // normalize and write y[b,e,qrow, h*64 + hs] (bf16 for proj GEMM)
#pragma unroll
  for (int r = 0; r < 8; r++) {
    int row = q0 + r + half * 8;
    float inv = 1.0f / l_i[r];
    __bf16* yr = yatt + ((size_t)((b * EE + e) * KSEL) + row) * CC + h * HS;
#pragma unroll
    for (int c = 0; c < 4; c++) yr[c * 16 + lr] = (__bf16)(Oacc[c][r] * inv);
  }
}

// ---------------------------------------------------------------------------
// out = x (residual stream base), 16 B per thread
// ---------------------------------------------------------------------------
__global__ __launch_bounds__(1024) void copy_kernel(const float4* __restrict__ x,
                                                    float4* __restrict__ out, int n4) {
  int i = blockIdx.x * 1024 + threadIdx.x;
  if (i < n4) out[i] = x[i];
}

// ---------------------------------------------------------------------------
// Gate softmax (E=2), combine experts, apply router weight, scatter-add
// (indices unique per batch -> plain stores)
// ---------------------------------------------------------------------------
__global__ __launch_bounds__(256) void combine_kernel(const float* __restrict__ x,
                                                      const int* __restrict__ idx,
                                                      const float* __restrict__ wsel,
                                                      const float* __restrict__ Wg,
                                                      const float* __restrict__ yproj,
                                                      float* __restrict__ out) {
  int row = blockIdx.x;  // b*KSEL + j
  int b = row >> 10, j = row & (KSEL - 1);
  int t = idx[row];
  const float* xr = x + ((size_t)b * TT + t) * CC;

  __shared__ float redg[512];
  float g0 = 0.f, g1 = 0.f;
  for (int c = threadIdx.x; c < CC; c += 256) {
    float v = xr[c];
    g0 += v * Wg[c * 2 + 0];
    g1 += v * Wg[c * 2 + 1];
  }
  redg[threadIdx.x]       = g0;
  redg[256 + threadIdx.x] = g1;
  __syncthreads();
  for (int s = 128; s > 0; s >>= 1) {
    if (threadIdx.x < s) {
      redg[threadIdx.x]       += redg[threadIdx.x + s];
      redg[256 + threadIdx.x] += redg[256 + threadIdx.x + s];
    }
    __syncthreads();
  }
  g0 = redg[0];
  g1 = redg[256];
  float mg = fmaxf(g0, g1);
  float e0 = expf(g0 - mg), e1 = expf(g1 - mg);
  float inv = 1.0f / (e0 + e1);
  float gt0 = e0 * inv, gt1 = e1 * inv;
  float wj = wsel[row];

  const float* y0 = yproj + ((size_t)(b * EE + 0) * KSEL + j) * CC;
  const float* y1 = yproj + ((size_t)(b * EE + 1) * KSEL + j) * CC;
  float* orow = out + ((size_t)b * TT + t) * CC;
  for (int c = threadIdx.x; c < CC; c += 256)
    orow[c] = xr[c] + (gt0 * y0[c] + gt1 * y1[c]) * wj;
}

// ---------------------------------------------------------------------------
// BCE loss: mean over B*T, single block (deterministic)
// ---------------------------------------------------------------------------
__global__ __launch_bounds__(1024) void loss_kernel(const float* __restrict__ pl,
                                                    const float* __restrict__ maskf,
                                                    float* __restrict__ out_loss) {
  __shared__ float red[1024];
  float s = 0.f;
  for (int i = threadIdx.x; i < BB * TT; i += 1024) {
    float p = pl[i], t = maskf[i];
    s += fmaxf(p, 0.f) - p * t + log1pf(expf(-fabsf(p)));
  }
  red[threadIdx.x] = s;
  __syncthreads();
  for (int k = 512; k > 0; k >>= 1) {
    if (threadIdx.x < k) red[threadIdx.x] += red[threadIdx.x + k];
    __syncthreads();
  }
  if (threadIdx.x == 0) out_loss[0] = red[0] / (float)(BB * TT);
}

// ---------------------------------------------------------------------------
extern "C" void kernel_launch(void* const* d_in, const int* in_sizes, int n_in,
                              void* d_out, int out_size, void* d_ws, size_t ws_size,
                              hipStream_t stream) {
  (void)in_sizes; (void)n_in; (void)ws_size;
  const float* x   = (const float*)d_in[0];
  const float* Wr  = (const float*)d_in[1];
  const float* Wq  = (const float*)d_in[2];
  const float* Wkv = (const float*)d_in[3];
  const float* Wp  = (const float*)d_in[4];
  const float* lnw = (const float*)d_in[5];
  const float* Wg  = (const float*)d_in[6];
  const float* Wpr = (const float*)d_in[7];

  float* out   = (float*)d_out;
  float* lossp = (float*)d_out + (out_size - 1);

  // workspace carve (256B-aligned offsets; y_att aliases q_f32, y_proj aliases kv_f32)
  char* ws = (char*)d_ws;
  float*  logits = (float*)(ws + 0);                           //  16 KB
  float*  pl     = (float*)(ws + 16384);                       //  16 KB
  float*  maskf  = (float*)(ws + 32768);                       //  16 KB
  int*    idx    = (int*)  (ws + 49152);                       //   8 KB
  float*  wsel   = (float*)(ws + 57344);                       //   8 KB
  __bf16* Wq16   = (__bf16*)(ws + 65536);                      //   1 MB
  __bf16* Wkv16  = (__bf16*)(ws + 1114112);                    //   1 MB
  __bf16* Wp16   = (__bf16*)(ws + 2162688);                    // 512 KB
  __bf16* sel16  = (__bf16*)(ws + 2686976);                    //   2 MB
  float*  qf     = (float*)(ws + 4784128);                     //   8 MB
  float*  kvf    = (float*)(ws + 13172736);                    //   8 MB
  __bf16* qh     = (__bf16*)(ws + 21561344);                   //   4 MB
  __bf16* kh     = (__bf16*)(ws + 25755648);                   //   2 MB
  __bf16* vh     = (__bf16*)(ws + 27852800);                   //   2 MB  -> 29,949,952 B total
  __bf16* yatt   = (__bf16*)qf;    // reuse (q_f32 dead after q_post)
  float*  yproj  = kvf;            // reuse (kv_f32 dead after kv_post)

  // 1) weights -> bf16
  tobf16_kernel<<<(CC * EE * CC + 255) / 256, 256, 0, stream>>>(Wq, Wq16, CC * EE * CC);
  tobf16_kernel<<<(CC * 2 * CC + 255) / 256, 256, 0, stream>>>(Wkv, Wkv16, CC * 2 * CC);
  tobf16_kernel<<<(CC * CC + 255) / 256, 256, 0, stream>>>(Wp, Wp16, CC * CC);

  // 2) router/pred logits + mask clear
  router_kernel<<<(BB * TT) / 8, 256, 0, stream>>>(x, Wr, Wpr, logits, pl, maskf);

  // 3) per-batch top-k
  topk_kernel<<<BB, 1024, 0, stream>>>(logits, idx, wsel, maskf);

  // 4) gather selected tokens (bf16)
  gather_kernel<<<BB * KSEL, 128, 0, stream>>>(x, idx, sel16);

  // 5) q = sel @ W_q   [2048 x 512] x [512 x 1024]
  gemm_bf16_kernel<<<(2048 / 16) * (1024 / 16) / 8, 256, 0, stream>>>(
      sel16, Wq16, qf, 2048, 1024, 512);
  // 6) kv = sel @ W_kv [2048 x 512] x [512 x 1024]
  gemm_bf16_kernel<<<(2048 / 16) * (1024 / 16) / 8, 256, 0, stream>>>(
      sel16, Wkv16, kvf, 2048, 1024, 512);

  // 7/8) RMSNorm + RoPE, head-major bf16 repack
  q_post_kernel<<<BB * EE * KSEL, 128, 0, stream>>>(qf, lnw, qh);
  kv_post_kernel<<<BB * KSEL, 128, 0, stream>>>(kvf, lnw, kh, vh);

  // 9) flash attention (2048 waves: B*E*NH*64 query blocks)
  attn_kernel<<<BB * EE * NHH * (KSEL / 16), 32, 0, stream>>>(qh, kh, vh, yatt);

  // 10) y @ W_proj [4096 x 512] x [512 x 512]
  gemm_bf16_kernel<<<(4096 / 16) * (512 / 16) / 8, 256, 0, stream>>>(
      yatt, Wp16, yproj, 4096, 512, 512);

  // 11) residual base, 12) gated combine + scatter
  copy_kernel<<<(BB * TT * CC / 4) / 1024, 1024, 0, stream>>>(
      (const float4*)x, (float4*)out, BB * TT * CC / 4);
  combine_kernel<<<BB * KSEL, 256, 0, stream>>>(x, idx, wsel, Wg, yproj, out);

  // 13) BCE loss
  loss_kernel<<<1, 1024, 0, stream>>>(pl, maskf, lossp);
}
